// GDAttention_5325759447379
// MI455X (gfx1250) — matrix operand order repaired and checked by reading, not compiled, not run on hardware
//
#include <hip/hip_runtime.h>

// Problem constants (reference: B=2, S=4096, D=128, H=8)
#define BB 2
#define SS 4096
#define DD 128
#define HH 8
#define HD (HH * DD)  // 1024

typedef __attribute__((ext_vector_type(16))) __bf16 v16bf;
typedef __attribute__((ext_vector_type(8)))  float  v8f;

// scale * log2(e): scores come out of QK^T already in base-2 domain
#define QSCALE (0.08838834764831845f * 1.4426950408889634f)  // 1/sqrt(128)*log2e
#define CLIP2  (10.0f * 1.4426950408889634f)                 // 10*log2e

__device__ __forceinline__ __bf16 f2bf(float f) { return (__bf16)f; }

// A-operand (16x32, MxK) loader. rowp points at [m][0] of the 16x32 chunk for
// THIS lane's row m = lane%16.  ISA 7.12.2 16-bit A layout: lane half hf holds
// K = hf*8+{0..7} in elems 0-7 and K = 16+hf*8+{0..7} in elems 8-15.
__device__ __forceinline__ v16bf load_a16x32(const __bf16* rowp) {
  const int hf = (threadIdx.x >> 4) & 1;
  v16bf a;
#pragma unroll
  for (int e = 0; e < 8; ++e) {
    a[e]     = rowp[hf * 8 + e];
    a[e + 8] = rowp[16 + hf * 8 + e];
  }
  return a;
}

// B-operand (32x16, KxN) loader. colp points at k=0 of the contiguous K-major
// vector for THIS lane's column n = lane%16. Lane half hf holds K = hf*8+{0..7}
// in elems 0-7 and K = 16+hf*8+{0..7} in elems 8-15.
__device__ __forceinline__ v16bf load_b32x16(const __bf16* colp) {
  const int hf = (threadIdx.x >> 4) & 1;
  v16bf b;
#pragma unroll
  for (int e = 0; e < 8; ++e) {
    b[e]     = colp[hf * 8 + e];
    b[e + 8] = colp[16 + hf * 8 + e];
  }
  return b;
}

// ---------------------------------------------------------------------------
// K0: convert x -> bf16 (row-major) and x^T -> bf16 [B, D, S] (V operand)
// ---------------------------------------------------------------------------
__global__ void cvt_x_kernel(const float* __restrict__ x,
                             __bf16* __restrict__ xbf,
                             __bf16* __restrict__ xT) {
  int i = blockIdx.x * blockDim.x + threadIdx.x;
  if (i >= BB * SS * DD) return;
  float v = x[i];
  __bf16 bv = f2bf(v);
  xbf[i] = bv;
  int d  = i & (DD - 1);
  int bs = i >> 7;            // i / 128
  int s  = bs & (SS - 1);
  int b  = bs >> 12;          // / 4096
  xT[((size_t)b * DD + d) * SS + s] = bv;
}

// K0b: W_q/W_k -> bf16 transposed [h, e, d]; W_o -> bf16 row-major [e, f]
// (H*D*D == 128*1024 == 131072, so one index covers all three)
__global__ void cvt_w_kernel(const float* __restrict__ Wq,
                             const float* __restrict__ Wk,
                             const float* __restrict__ Wo,
                             __bf16* __restrict__ WqT,
                             __bf16* __restrict__ WkT,
                             __bf16* __restrict__ Wob) {
  int i = blockIdx.x * blockDim.x + threadIdx.x;
  if (i >= HH * DD * DD) return;
  int e  = i & (DD - 1);
  int hd = i >> 7;
  int d  = hd & (DD - 1);
  int h  = hd >> 7;
  size_t ti = (((size_t)h * DD + e) * DD) + d;
  WqT[ti] = f2bf(Wq[i]);
  WkT[ti] = f2bf(Wk[i]);
  Wob[i]  = f2bf(Wo[i]);
}

// ---------------------------------------------------------------------------
// K1: Q/K projection.  One wave computes a 16(S) x 128(E) tile of both Q and K
// for one (b, h).  A = x rows (shared), B = W^T[h, e, :] contiguous columns.
// (1/sqrt(D))*log2(e) is folded into Q: QK^T then yields base-2-domain scores
// (clipping commutes with the positive linear rescale, so this is exact).
// ---------------------------------------------------------------------------
__global__ void __launch_bounds__(32)
qk_proj_kernel(const __bf16* __restrict__ xbf,
               const __bf16* __restrict__ WqT,
               const __bf16* __restrict__ WkT,
               __bf16* __restrict__ Qb,
               __bf16* __restrict__ Kb) {
  const int st = blockIdx.x, h = blockIdx.y, b = blockIdx.z;
  const int lane = threadIdx.x;
  const int m  = lane & 15;
  const int hf = lane >> 4;

  const __bf16* xrow = xbf + ((size_t)b * SS + st * 16 + m) * DD;
  v16bf a[4];
#pragma unroll
  for (int c = 0; c < 4; ++c) a[c] = load_a16x32(xrow + c * 32);

  const size_t wbase   = (size_t)h * DD * DD;
  const size_t rowbase = ((size_t)(b * HH + h) * SS + st * 16);

#pragma unroll
  for (int t = 0; t < 8; ++t) {
    v8f cq, ck;
#pragma unroll
    for (int r = 0; r < 8; ++r) { cq[r] = 0.0f; ck[r] = 0.0f; }
    const __bf16* wq_col = WqT + wbase + (size_t)(t * 16 + m) * DD;
    const __bf16* wk_col = WkT + wbase + (size_t)(t * 16 + m) * DD;
#pragma unroll
    for (int c = 0; c < 4; ++c) {
      v16bf bq = load_b32x16(wq_col + c * 32);
      cq = __builtin_amdgcn_wmma_f32_16x16x32_bf16(false, a[c], false, bq,
                                                   (short)0, cq, false, false);
      v16bf bk = load_b32x16(wk_col + c * 32);
      ck = __builtin_amdgcn_wmma_f32_16x16x32_bf16(false, a[c], false, bk,
                                                   (short)0, ck, false, false);
    }
#pragma unroll
    for (int r = 0; r < 8; ++r) {
      int M = r + hf * 8;
      Qb[(rowbase + M) * DD + t * 16 + m] = f2bf(cq[r] * QSCALE);
      Kb[(rowbase + M) * DD + t * 16 + m] = f2bf(ck[r]);
    }
  }
}

// ---------------------------------------------------------------------------
// K2: causal attention.  One wave owns 16 query rows of one (b, h).
// Because the reference clips scores to +-10 BEFORE softmax, the exponent
// domain is bounded: a FIXED shift of 10 makes exp(s-10) in [e^-20, 1], so no
// running max / rescale is needed (softmax is shift-invariant -> exact).
// Row sums are pure accumulations: each lane keeps private partials for its
// own columns; the 16-lane shuffle reduction happens once at the end.
// Per 32-key strip: 8 score WMMAs -> clamp/mask/exp2 in C/D register layout
// -> P through LDS (bf16, C/D->A transpose only) -> 8 P*V WMMAs against xT.
// ---------------------------------------------------------------------------
__global__ void __launch_bounds__(32)
attn_kernel(const __bf16* __restrict__ Qb,
            const __bf16* __restrict__ Kb,
            const __bf16* __restrict__ xT,
            __bf16* __restrict__ Yb) {
  __shared__ __bf16 sP[16 * 32];

  const int qt = blockIdx.x, h = blockIdx.y, b = blockIdx.z;
  const int lane = threadIdx.x;
  const int n  = lane & 15;   // column / row index within 16-lane half
  const int hf = lane >> 4;
  const int q0 = qt * 16;
  const size_t bh = (size_t)b * HH + h;

  // Q tile (16x128, pre-scaled to base-2 domain) in A-layout registers.
  const __bf16* qrow = Qb + (bh * SS + (size_t)(q0 + n)) * DD;
  v16bf aq[4];
#pragma unroll
  for (int c = 0; c < 4; ++c) aq[c] = load_a16x32(qrow + c * 32);

  v8f acc[8];
#pragma unroll
  for (int t = 0; t < 8; ++t)
#pragma unroll
    for (int r = 0; r < 8; ++r) acc[t][r] = 0.0f;

  // Per-lane PARTIAL row sums (this lane's columns only) for rows r + hf*8.
  float lpart[8];
#pragma unroll
  for (int r = 0; r < 8; ++r) lpart[r] = 0.0f;

  for (int k0 = 0; k0 < q0 + 16; k0 += 32) {
    // ---- scores: two 16x16 tiles across this 32-key strip ----
    v8f sc0, sc1;
#pragma unroll
    for (int r = 0; r < 8; ++r) { sc0[r] = 0.0f; sc1[r] = 0.0f; }
    const __bf16* krow0 = Kb + (bh * SS + (size_t)(k0 + n)) * DD;
    const __bf16* krow1 = krow0 + (size_t)16 * DD;
    // Prefetch next strip's K rows behind this strip's WMMAs.
    if (k0 + 32 < q0 + 16) {
      __builtin_prefetch(krow0 + (size_t)32 * DD, 0, 0);
      __builtin_prefetch(krow1 + (size_t)32 * DD, 0, 0);
    }
#pragma unroll
    for (int c = 0; c < 4; ++c) {
      v16bf bk0 = load_b32x16(krow0 + c * 32);
      sc0 = __builtin_amdgcn_wmma_f32_16x16x32_bf16(false, aq[c], false, bk0,
                                                    (short)0, sc0, false, false);
      v16bf bk1 = load_b32x16(krow1 + c * 32);
      sc1 = __builtin_amdgcn_wmma_f32_16x16x32_bf16(false, aq[c], false, bk1,
                                                    (short)0, sc1, false, false);
    }

    // ---- fixed-shift softmax numerator, C/D layout, all 32 lanes ----
    const int col0 = k0 + n;
    const int col1 = k0 + 16 + n;
#pragma unroll
    for (int r = 0; r < 8; ++r) {
      const int q = q0 + r + hf * 8;                 // this element's query row
      float s0 = fminf(fmaxf(sc0[r], -CLIP2), CLIP2);
      float s1 = fminf(fmaxf(sc1[r], -CLIP2), CLIP2);
      s0 = (col0 <= q) ? s0 : -1e30f;                // causal mask
      s1 = (col1 <= q) ? s1 : -1e30f;
      float p0 = exp2f(s0 - CLIP2);                  // in [2^-28.9, 1]; masked->0
      float p1 = exp2f(s1 - CLIP2);
      lpart[r] += p0 + p1;
      // Stage P (bf16) to LDS for the C/D -> A layout transpose.
      sP[(r + hf * 8) * 32 + n]      = f2bf(p0);
      sP[(r + hf * 8) * 32 + 16 + n] = f2bf(p1);
    }
    __syncthreads();

    // ---- P (16x32) @ V (32x128), V = x^T columns; no rescale needed ----
    v16bf pa = load_a16x32(sP + n * 32);
#pragma unroll
    for (int t = 0; t < 8; ++t) {
      const __bf16* vcol = xT + ((size_t)b * DD + (t * 16 + n)) * SS + k0;
      v16bf bv = load_b32x16(vcol);
      acc[t] = __builtin_amdgcn_wmma_f32_16x16x32_bf16(false, pa, false, bv,
                                                       (short)0, acc[t], false, false);
    }
    __syncthreads();  // P reads done before next strip rewrites sP
  }

  // ---- one-time row-sum reduction (xor 1,2,4,8 stays in each half) ----
  float inv[8];
#pragma unroll
  for (int r = 0; r < 8; ++r) {
    float l = lpart[r];
    l += __shfl_xor(l, 1, 32);
    l += __shfl_xor(l, 2, 32);
    l += __shfl_xor(l, 4, 32);
    l += __shfl_xor(l, 8, 32);
    inv[r] = 1.0f / l;
  }

  // ---- finalize: divide by row sums, write Y as [B, S, H*D] bf16 ----
#pragma unroll
  for (int t = 0; t < 8; ++t) {
#pragma unroll
    for (int r = 0; r < 8; ++r) {
      int M = r + hf * 8;
      Yb[((size_t)b * SS + q0 + M) * HD + (size_t)h * DD + t * 16 + n] =
          f2bf(acc[t][r] * inv[r]);
    }
  }
}

// ---------------------------------------------------------------------------
// K3: output projection  out[B*S,128] = Y[B*S,1024] @ Wo[128,1024]^T  (fp32 out)
// ---------------------------------------------------------------------------
__global__ void __launch_bounds__(32)
oproj_kernel(const __bf16* __restrict__ Yb,
             const __bf16* __restrict__ Wob,
             float* __restrict__ out) {
  const int mt = blockIdx.x;
  const int lane = threadIdx.x;
  const int m  = lane & 15;
  const int hf = lane >> 4;

  const __bf16* yrow = Yb + ((size_t)mt * 16 + m) * HD;

  v8f acc[8];
#pragma unroll
  for (int t = 0; t < 8; ++t)
#pragma unroll
    for (int r = 0; r < 8; ++r) acc[t][r] = 0.0f;

  for (int c = 0; c < HD / 32; ++c) {   // 32 K-chunks over f = 1024
    v16bf a = load_a16x32(yrow + c * 32);
#pragma unroll
    for (int t = 0; t < 8; ++t) {
      const __bf16* wcol = Wob + (size_t)(t * 16 + m) * HD + c * 32;
      v16bf bv = load_b32x16(wcol);
      acc[t] = __builtin_amdgcn_wmma_f32_16x16x32_bf16(false, a, false, bv,
                                                       (short)0, acc[t], false, false);
    }
  }

#pragma unroll
  for (int t = 0; t < 8; ++t)
#pragma unroll
    for (int r = 0; r < 8; ++r)
      out[((size_t)mt * 16 + r + hf * 8) * DD + t * 16 + m] = acc[t][r];
}

// ---------------------------------------------------------------------------
extern "C" void kernel_launch(void* const* d_in, const int* in_sizes, int n_in,
                              void* d_out, int out_size, void* d_ws, size_t ws_size,
                              hipStream_t stream) {
  const float* x  = (const float*)d_in[0];
  const float* Wq = (const float*)d_in[1];
  const float* Wk = (const float*)d_in[2];
  const float* Wo = (const float*)d_in[3];
  float* out = (float*)d_out;

  char* ws = (char*)d_ws;
  size_t off = 0;
  auto carve = [&](size_t count) -> __bf16* {
    __bf16* p = (__bf16*)(ws + off);
    off += (count * sizeof(__bf16) + 255) & ~(size_t)255;
    return p;
  };
  __bf16* xbf = carve((size_t)BB * SS * DD);       // x, bf16 row-major
  __bf16* xT  = carve((size_t)BB * DD * SS);       // x^T per batch (V operand)
  __bf16* WqT = carve((size_t)HH * DD * DD);       // W_q transposed [h,e,d]
  __bf16* WkT = carve((size_t)HH * DD * DD);
  __bf16* Wob = carve((size_t)DD * HD);            // W_o row-major bf16
  __bf16* Qb  = carve((size_t)BB * HH * SS * DD);  // Q [b,h,s,e] (pre-scaled)
  __bf16* Kb  = carve((size_t)BB * HH * SS * DD);  // K [b,h,s,e]
  __bf16* Yb  = carve((size_t)BB * SS * HD);       // attn out [b,s,h*d]
  (void)ws_size; (void)in_sizes; (void)n_in; (void)out_size;

  cvt_x_kernel<<<dim3((BB * SS * DD + 255) / 256), dim3(256), 0, stream>>>(x, xbf, xT);
  cvt_w_kernel<<<dim3((HH * DD * DD + 255) / 256), dim3(256), 0, stream>>>(
      Wq, Wk, Wo, WqT, WkT, Wob);
  qk_proj_kernel<<<dim3(SS / 16, HH, BB), dim3(32), 0, stream>>>(xbf, WqT, WkT, Qb, Kb);
  attn_kernel<<<dim3(SS / 16, HH, BB), dim3(32), 0, stream>>>(Qb, Kb, xT, Yb);
  oproj_kernel<<<dim3(BB * SS / 16), dim3(32), 0, stream>>>(Yb, Wob, out);
}
